// TypedBinaryTreeLSTM_79783312490658
// MI455X (gfx1250) — matrix operand order repaired and checked by reading, not compiled, not run on hardware
//
#include <hip/hip_runtime.h>

// ---------------- problem constants (from reference) ----------------
#define HV   128
#define HT   17
#define KT   8          // number of parent types (K in reference)
#define DD   290        // 2*(HV+HT)
#define DPAD 320        // DD padded to multiple of 32 for WMMA K
#define OO   640        // 5*HV gate outputs

typedef __attribute__((ext_vector_type(16))) __bf16 v16bf;
typedef __attribute__((ext_vector_type(8)))  __bf16 v8bf;
typedef __attribute__((ext_vector_type(8)))  float  v8f;

__device__ __forceinline__ float sigf(float x) {
    return 1.0f / (1.0f + __expf(-x));
}
__device__ __forceinline__ float tanh_fast(float x) {
    // tanh via exp, safe for large |x|
    float t = __expf(-2.0f * fabsf(x));   // t in (0,1]
    float r = (1.0f - t) / (1.0f + t);
    return copysignf(r, x);
}

// ---------------- prep: W (K,O,D) f32 -> bf16 padded (K,O,DPAD) ----------------
__global__ void prep_w_kernel(const float* __restrict__ W, __bf16* __restrict__ Wbf, int total) {
    int i = blockIdx.x * blockDim.x + threadIdx.x;
    if (i >= total) return;
    int d  = i % DPAD;
    int ko = i / DPAD;                     // k*OO + o
    float v = (d < DD) ? W[(size_t)ko * DD + d] : 0.0f;
    Wbf[i] = (__bf16)v;
}

// ---------------- bucketing by parent_type ----------------
__global__ void init_meta_kernel(int* __restrict__ perm, int permCount, int* __restrict__ counts) {
    int i = blockIdx.x * blockDim.x + threadIdx.x;
    if (i < permCount) perm[i] = -1;
    if (i < KT) counts[i] = 0;
}

__global__ void histo_kernel(const int* __restrict__ parent_type, int* __restrict__ counts, int N) {
    int i = blockIdx.x * blockDim.x + threadIdx.x;
    if (i < N) atomicAdd(&counts[parent_type[i]], 1);
}

__global__ void offsets_kernel(const int* __restrict__ counts, int* __restrict__ cursors) {
    if (blockIdx.x == 0 && threadIdx.x == 0) {
        int acc = 0;
        for (int k = 0; k < KT; ++k) {
            cursors[k] = acc;
            acc += ((counts[k] + 63) >> 6) << 6;   // pad each type segment to 64
        }
    }
}

__global__ void scatter_kernel(const int* __restrict__ parent_type, int* __restrict__ cursors,
                               int* __restrict__ perm, int N) {
    int i = blockIdx.x * blockDim.x + threadIdx.x;
    if (i < N) {
        int t   = parent_type[i];
        int pos = atomicAdd(&cursors[t], 1);
        perm[pos] = i;
    }
}

// ---------------- main fused gather + async-LDS + WMMA GEMM + LSTM epilogue ----------------
// Block: 256 threads (8 waves), 64 nodes (one type), full 640 gate outputs.
// Per outer iteration t, the block async-stages the 160 W rows needed by the two
// active h-groups (2t, 2t+1) into LDS; all 4 M-subtile waves share that stage.
__global__ __launch_bounds__(256)
void tree_lstm_main(const float* __restrict__ h_pool, const float* __restrict__ c_pool,
                    const float* __restrict__ t_pool, const int* __restrict__ child_idx,
                    const int* __restrict__ parent_type, const float* __restrict__ b,
                    const __bf16* __restrict__ Wbf, const int* __restrict__ perm,
                    float* __restrict__ out, int permCount)
{
    __shared__ __attribute__((aligned(16))) __bf16 As[64  * DPAD];   // 40 KB activations
    __shared__ __attribute__((aligned(16))) __bf16 Bs[160 * DPAD];   // 100 KB weight stage
    __shared__ int nodeId[64], liS[64], riS[64];
    __shared__ int typeS;

    const int tid  = threadIdx.x;
    const int base = blockIdx.x * 64;

    if (tid < 64) {
        int idx = base + tid;
        int p = (idx < permCount) ? perm[idx] : -1;
        nodeId[tid] = p;
        int li = 0, ri = 0;
        if (p >= 0) { li = child_idx[2 * p]; ri = child_idx[2 * p + 1]; }
        liS[tid] = li; riS[tid] = ri;
    }
    __syncthreads();
    if (tid == 0) {
        int k = -1;
        for (int i = 0; i < 64; ++i) {
            int p = nodeId[i];
            if (p >= 0) { k = parent_type[p]; break; }
        }
        typeS = k;
    }
    __syncthreads();
    const int ktype = typeS;
    if (ktype < 0) return;                       // uniform exit: all-padding tile

    // Gather x = [h[li] | t[li] | h[ri] | t[ri] | 0-pad] into LDS as bf16.
    for (int e = tid; e < 64 * DPAD; e += 256) {
        int row = e / DPAD, d = e - row * DPAD;
        float v = 0.0f;
        if (nodeId[row] >= 0) {
            int li = liS[row], ri = riS[row];
            if      (d < HV)          v = h_pool[(size_t)li * HV + d];
            else if (d < HV + HT)     v = t_pool[(size_t)li * HT + (d - HV)];
            else if (d < 2*HV + HT)   v = h_pool[(size_t)ri * HV + (d - (HV + HT))];
            else if (d < DD)          v = t_pool[(size_t)ri * HT + (d - (2*HV + HT))];
        }
        As[e] = (__bf16)v;
    }
    __syncthreads();

    const int wave   = tid >> 5;
    const int lane   = tid & 31;
    const int msub   = wave & 3;                 // 16-node row group
    const int parity = wave >> 2;                // which h-group of the staged pair
    const int lrow   = lane & 15;
    const int hiHalf = lane >> 4;
    const int aoff   = hiHalf * 8;               // A fragment: K = aoff+{0..7} and aoff+{16..23}
    const int boff   = hiHalf * 8;               // B fragment: K = boff+{0..7} and boff+{16..23}
    const int arow   = msub * 16 + lrow;

    const __bf16* Wk = Wbf + (size_t)ktype * OO * DPAD;
    const float*  bk = b   + ktype * OO;
    const unsigned bsBase = (unsigned)(uintptr_t)(&Bs[0]);   // LDS byte offset of Bs

    for (int t = 0; t < 4; ++t) {
        // ---- async-stage 160 W rows (parities {0,1} x 5 gates x 16 cols) into Bs ----
        // 160 rows * 640 B = 6400 chunks of 16 B; 25 chunks per thread.
        for (int c = tid; c < 6400; c += 256) {
            int row   = c / 40;                  // Bs row 0..159 (40 x 16B chunks per row)
            int off16 = c - row * 40;
            int p   = row / 80;
            int rem = row - p * 80;
            int g   = rem >> 4;
            int r   = rem & 15;
            int n   = g * HV + (2 * t + p) * 16 + r;     // W output row
            unsigned long long gsrc =
                (unsigned long long)(uintptr_t)(Wk + (size_t)n * DPAD) +
                (unsigned)(off16 * 16);
            unsigned ldst = bsBase + (unsigned)(row * 640 + off16 * 16);
            asm volatile("global_load_async_to_lds_b128 %0, %1, off"
                         :: "v"(ldst), "v"(gsrc) : "memory");
        }
        asm volatile("s_wait_asynccnt 0x0" ::: "memory");  // my transfers landed
        __syncthreads();                                    // everyone's landed

        const int hg = 2 * t + parity;
        const int j  = hg * 16 + lrow;           // h index 0..127 owned by this lane column
        v8f acc[5] = {};                         // i, fl, fr, u, o accumulators (f32)

        #pragma unroll
        for (int kb = 0; kb < DPAD; kb += 32) {
            const __bf16* pA = As + (size_t)arow * DPAD + kb + aoff;
            v8bf alo = *(const v8bf*)(pA);
            v8bf ahi = *(const v8bf*)(pA + 16);
            v16bf afrag = __builtin_shufflevector(alo, ahi,
                0,1,2,3,4,5,6,7,8,9,10,11,12,13,14,15);
            #pragma unroll
            for (int g = 0; g < 5; ++g) {
                const __bf16* pB = Bs + (size_t)(parity * 80 + g * 16 + lrow) * DPAD + kb + boff;
                v8bf blo = *(const v8bf*)(pB);
                v8bf bhi = *(const v8bf*)(pB + 16);
                v16bf bfrag = __builtin_shufflevector(blo, bhi,
                    0,1,2,3,4,5,6,7,8,9,10,11,12,13,14,15);
                acc[g] = __builtin_amdgcn_wmma_f32_16x16x32_bf16(
                    false, afrag, false, bfrag, (short)0, acc[g], false, false);
            }
        }

        // LSTM epilogue: c = sig(i)*tanh(u) + sig(fl)*cl + sig(fr)*cr ; h = sig(o)*tanh(c)
        float bi  = bk[j];
        float bfl = bk[HV   + j];
        float bfr = bk[2*HV + j];
        float bu  = bk[3*HV + j];
        float bo  = bk[4*HV + j];
        #pragma unroll
        for (int v = 0; v < 8; ++v) {
            int row  = msub * 16 + v + hiHalf * 8;   // C/D layout: M = vgpr + 8*(lane>=16)
            int node = nodeId[row];
            if (node < 0) continue;
            int li = liS[row], ri = riS[row];
            float cl = c_pool[(size_t)li * HV + j];
            float cr = c_pool[(size_t)ri * HV + j];
            float gi = acc[0][v] + bi;
            float gl = acc[1][v] + bfl;
            float gr = acc[2][v] + bfr;
            float gu = acc[3][v] + bu;
            float go = acc[4][v] + bo;
            float c  = sigf(gi) * tanh_fast(gu) + sigf(gl) * cl + sigf(gr) * cr;
            float h  = sigf(go) * tanh_fast(c);
            out[(size_t)node * 256 + j]       = h;
            out[(size_t)node * 256 + 128 + j] = c;
        }
        __syncthreads();                         // Bs reusable for next iteration
    }
}

// ---------------- launch ----------------
extern "C" void kernel_launch(void* const* d_in, const int* in_sizes, int n_in,
                              void* d_out, int out_size, void* d_ws, size_t ws_size,
                              hipStream_t stream) {
    const float* h_pool      = (const float*)d_in[0];
    const float* c_pool      = (const float*)d_in[1];
    const float* t_pool      = (const float*)d_in[2];
    const int*   child_idx   = (const int*)  d_in[3];
    const int*   parent_type = (const int*)  d_in[4];
    const float* W           = (const float*)d_in[5];
    const float* bvec        = (const float*)d_in[6];
    float*       out         = (float*)d_out;

    const int N = in_sizes[3] / 2;

    // workspace layout: [Wbf16 padded | perm | counts(8) | cursors(8)]
    char* ws = (char*)d_ws;
    __bf16* Wbf = (__bf16*)ws;
    size_t wbytes = ((size_t)KT * OO * DPAD * sizeof(__bf16) + 255) & ~(size_t)255;

    const int nBlocks   = (N + KT * 63 + 63) / 64;   // worst-case per-type padding
    const int permCount = nBlocks * 64;
    int* perm = (int*)(ws + wbytes);
    size_t pbytes = (((size_t)permCount * sizeof(int)) + 255) & ~(size_t)255;
    int* counts  = (int*)(ws + wbytes + pbytes);
    int* cursors = counts + KT;

    const int totalW = KT * OO * DPAD;
    prep_w_kernel   <<<(totalW    + 255) / 256, 256, 0, stream>>>(W, Wbf, totalW);
    init_meta_kernel<<<(permCount + 255) / 256, 256, 0, stream>>>(perm, permCount, counts);
    histo_kernel    <<<(N         + 255) / 256, 256, 0, stream>>>(parent_type, counts, N);
    offsets_kernel  <<<1, 1, 0, stream>>>(counts, cursors);
    scatter_kernel  <<<(N         + 255) / 256, 256, 0, stream>>>(parent_type, cursors, perm, N);
    tree_lstm_main  <<<nBlocks, 256, 0, stream>>>(h_pool, c_pool, t_pool, child_idx,
                                                  parent_type, bvec, Wbf, perm, out, permCount);
}
